// QATLinear_45157286150363
// MI455X (gfx1250) — compile-verified, matrix operand order checked by reading, not run on hardware
//
#include <hip/hip_runtime.h>

typedef __attribute__((ext_vector_type(16))) _Float16 v16h;
typedef __attribute__((ext_vector_type(8)))  _Float16 v8h;
typedef __attribute__((ext_vector_type(4)))  _Float16 v4h;
typedef __attribute__((ext_vector_type(8)))  float    v8f;

#define TILE_M 128
#define TILE_N 128
#define TILE_K 32
#define LDS_STRIDE 40   // halves: 80-byte row stride (16B aligned, bank-conflict pad)

// fp16 (E5M10) quantization: clamp to +-max_fp16 then RNE convert (denorms on).
__device__ __forceinline__ _Float16 fp16_quant(float v) {
    v = fminf(fmaxf(v, -65504.0f), 65504.0f);
    return (_Float16)v;   // v_cvt_f16_f32, round-to-nearest-even
}

__global__ __launch_bounds__(256, 2)
void qat_linear_wmma_kernel(const float* __restrict__ X,
                            const float* __restrict__ W,
                            const float* __restrict__ Bias,
                            float* __restrict__ Out,
                            int M, int N, int K) {
    __shared__ _Float16 Xs[2][TILE_M][LDS_STRIDE];
    __shared__ _Float16 Ws[2][TILE_N][LDS_STRIDE];

    const int tid  = threadIdx.x;
    const int lane = tid & 31;
    const int wave = tid >> 5;
    const int wm   = wave & 3;      // 4 wave-rows of 32 rows
    const int wn   = wave >> 2;     // 2 wave-cols of 64 cols

    const int blockM = blockIdx.y * TILE_M;
    const int blockN = blockIdx.x * TILE_N;

    // ---------------- global -> register tile staging ----------------
    float4 xg[4], wg[4];
    auto load_tile = [&](int k0) {
#pragma unroll
        for (int i = 0; i < 4; ++i) {
            int idx = tid + 256 * i;          // 1024 float4 chunks per matrix tile
            int row = idx >> 3;               // 8 float4 per 32-wide k row
            int kq  = (idx & 7) << 2;
            xg[i] = *(const float4*)(X + (size_t)(blockM + row) * K + k0 + kq);
            wg[i] = *(const float4*)(W + (size_t)(blockN + row) * K + k0 + kq);
        }
    };
    // ---------------- register -> LDS (quantize to fp16) ----------------
    auto store_tile = [&](int buf) {
#pragma unroll
        for (int i = 0; i < 4; ++i) {
            int idx = tid + 256 * i;
            int row = idx >> 3;
            int kq  = (idx & 7) << 2;
            v4h hx, hw;
            hx[0] = fp16_quant(xg[i].x); hx[1] = fp16_quant(xg[i].y);
            hx[2] = fp16_quant(xg[i].z); hx[3] = fp16_quant(xg[i].w);
            hw[0] = fp16_quant(wg[i].x); hw[1] = fp16_quant(wg[i].y);
            hw[2] = fp16_quant(wg[i].z); hw[3] = fp16_quant(wg[i].w);
            *(v4h*)&Xs[buf][row][kq] = hx;    // ds_store_b64
            *(v4h*)&Ws[buf][row][kq] = hw;
        }
    };

    // ---------------- accumulators ----------------
    v8f acc[2][4];
    const v8f vzero = {0.f, 0.f, 0.f, 0.f, 0.f, 0.f, 0.f, 0.f};
#pragma unroll
    for (int am = 0; am < 2; ++am)
#pragma unroll
        for (int bn = 0; bn < 4; ++bn)
            acc[am][bn] = vzero;

    // ISA 16-bit A/B fragment layout (wave32):
    //  lanes 0-15  hold K 0..7  and 16..23  (two b128 chunks)
    //  lanes 16-31 hold K 8..15 and 24..31
    const int half = lane & 15;
    const int koff = (lane >> 4) << 3;        // 0 or 8 halves

    load_tile(0);
    store_tile(0);
    __syncthreads();

    const int NK = K / TILE_K;
    int buf = 0;
    for (int kt = 0; kt < NK; ++kt) {
        if (kt + 1 < NK) load_tile((kt + 1) * TILE_K);   // prefetch next K-tile

        v16h afrag[2], bfrag[4];
#pragma unroll
        for (int am = 0; am < 2; ++am) {
            const _Float16* p = &Xs[buf][wm * 32 + am * 16 + half][koff];
            union { v16h v; v8h h[2]; } u;
            u.h[0] = *(const v8h*)p;          // ds_load_b128
            u.h[1] = *(const v8h*)(p + 16);   // ds_load_b128
            afrag[am] = u.v;
        }
#pragma unroll
        for (int bn = 0; bn < 4; ++bn) {
            const _Float16* p = &Ws[buf][wn * 64 + bn * 16 + half][koff];
            union { v16h v; v8h h[2]; } u;
            u.h[0] = *(const v8h*)p;
            u.h[1] = *(const v8h*)(p + 16);
            bfrag[bn] = u.v;
        }

#pragma unroll
        for (int am = 0; am < 2; ++am)
#pragma unroll
            for (int bn = 0; bn < 4; ++bn)
                acc[am][bn] = __builtin_amdgcn_wmma_f32_16x16x32_f16(
                    /*neg_a=*/false, afrag[am],
                    /*neg_b=*/false, bfrag[bn],
                    /*c_mod=*/(short)0, acc[am][bn],
                    /*reuse_a=*/false, /*reuse_b=*/false);

        if (kt + 1 < NK) store_tile(buf ^ 1);
        __syncthreads();
        buf ^= 1;
    }

    // ---------------- epilogue: + quantized bias, streaming stores ----------------
    float bq[4];
#pragma unroll
    for (int bn = 0; bn < 4; ++bn) {
        int col = blockN + wn * 64 + bn * 16 + half;
        bq[bn] = (float)fp16_quant(Bias[col]);
    }
    const int rowbase = (lane < 16) ? 0 : 8;  // C/D layout: VGPR r -> M=r / M=8+r
#pragma unroll
    for (int am = 0; am < 2; ++am) {
#pragma unroll
        for (int bn = 0; bn < 4; ++bn) {
            int col = blockN + wn * 64 + bn * 16 + half;
#pragma unroll
            for (int r = 0; r < 8; ++r) {
                int row = blockM + wm * 32 + am * 16 + rowbase + r;
                __builtin_nontemporal_store(acc[am][bn][r] + bq[bn],
                                            Out + (size_t)row * N + col);
            }
        }
    }
}

extern "C" void kernel_launch(void* const* d_in, const int* in_sizes, int n_in,
                              void* d_out, int out_size, void* d_ws, size_t ws_size,
                              hipStream_t stream) {
    const float* x    = (const float*)d_in[0];   // [M, K]
    const float* w    = (const float*)d_in[1];   // [N, K]
    const float* bias = (const float*)d_in[2];   // [N]
    float* out = (float*)d_out;                  // [M, N]

    const int N = in_sizes[2];                   // D_OUT
    const int K = in_sizes[1] / N;               // D_IN
    const int M = in_sizes[0] / K;               // rows of x

    dim3 grid(N / TILE_N, M / TILE_M);
    qat_linear_wmma_kernel<<<grid, 256, 0, stream>>>(x, w, bias, out, M, N, K);
}
